// SupernodePooling_14044543058266
// MI455X (gfx1250) — matrix operand order repaired and checked by reading, not compiled, not run on hardware
//
#include <hip/hip_runtime.h>

#define N_PTS 65536
#define M_SN  2048
#define KNB   32
#define CAP   64
#define DXF   16
#define HDIM  128
#define R2C   0.0025f   // 0.05^2
#define WV    4

typedef __attribute__((ext_vector_type(16))) _Float16 v16h;
typedef __attribute__((ext_vector_type(8)))  float    v8f;
typedef __attribute__((ext_vector_type(4)))  int      v4i;

// ---------------------------------------------------------------------------
// Optional CDNA5 fast paths, guarded so compilation can never break.
#if defined(__has_builtin)
#if __has_builtin(__builtin_amdgcn_global_load_async_to_lds_b128) && \
    __has_builtin(__builtin_amdgcn_s_wait_asynccnt)
#define HAVE_ASYNC_COPY 1
#endif
#if __has_builtin(__builtin_amdgcn_ballot_w32)
#define HAVE_BALLOT32 1
#endif
#endif

typedef __attribute__((address_space(1))) v4i ga_v4i;   // global (prints as __device__)
typedef __attribute__((address_space(3))) v4i ls_v4i;   // LDS    (prints as __shared__)

__device__ __forceinline__ void copy16_to_lds(void* lds_dst, const void* g_src) {
#if defined(HAVE_ASYNC_COPY)
    ga_v4i* gp = (ga_v4i*)(unsigned long long)(uintptr_t)g_src;
    ls_v4i* lp = (ls_v4i*)(unsigned)(uintptr_t)lds_dst;  // low 32b of flat addr == LDS offset
    __builtin_amdgcn_global_load_async_to_lds_b128(gp, lp, 0, 0);
#else
    *(uint4*)lds_dst = *(const uint4*)g_src;
#endif
}
__device__ __forceinline__ void async_copy_fence() {
#if defined(HAVE_ASYNC_COPY)
    __builtin_amdgcn_s_wait_asynccnt(0);
#endif
}

// ---------------------------------------------------------------------------
// A-fragment loader (16x32 f16 tile, row-major, row stride ld halves).
// lane L: row=L&15, half=L>>4; VGPR i holds K pair k=half*16+i*2
// -> two contiguous 16B runs per lane, compiler emits ds_load_b128 x2.
__device__ __forceinline__ v16h load_frag_a(const _Float16* base, int ld, int lane) {
    const int row  = lane & 15;
    const int half = lane >> 4;
    v16h a = {};
#pragma unroll
    for (int i = 0; i < 8; ++i) {
        const int k = half * 16 + i * 2;
        a[2 * i]     = base[row * ld + k];
        a[2 * i + 1] = base[row * ld + k + 1];
    }
    return a;
}

// Branchless gelu (tanh approximation, matching jax.nn.gelu approximate=True)
__device__ __forceinline__ float gelu_f(float v) {
    const float u = 0.7978845608028654f * (v + 0.044715f * v * v * v);
#if defined(__has_builtin) && __has_builtin(__builtin_amdgcn_tanhf)
    const float t = __builtin_amdgcn_tanhf(u);          // v_tanh_f32
#else
    const float uc = fminf(fmaxf(u, -9.0f), 9.0f);
    const float e  = __expf(-2.0f * uc);                 // v_exp_f32, no branches
    const float t  = (1.0f - e) / (1.0f + e);
#endif
    return 0.5f * v * (1.0f + t);
}

// ---------------------------------------------------------------------------
// Kernel 0: build fragment-swizzled f16 weights.
// B-fragment element e of lane L for col-tile nt (and row-tile kt for W2):
//   k = (L>>4)*16 + e ,  col = nt*16 + (L&15)
// w1f: [8 nt][32 lane][16 e]   (W1 padded 19->32 K rows)
// w2f: [4 kt][8 nt][32 lane][16 e], stored contiguously after w1f.
__global__ __launch_bounds__(256) void prep_weights(const float* __restrict__ W1,
                                                    const float* __restrict__ W2,
                                                    _Float16* __restrict__ wfrag) {
    const int t = blockIdx.x * 256 + threadIdx.x;
    if (t < 8 * 32 * 16) {
        const int e = t & 15, lane = (t >> 4) & 31, nt = t >> 9;
        const int k = (lane >> 4) * 16 + e;
        const int c = nt * 16 + (lane & 15);
        wfrag[t] = (k < DXF + 3) ? (_Float16)W1[k * HDIM + c] : (_Float16)0.0f;
    } else if (t < 8 * 32 * 16 + 4 * 8 * 32 * 16) {
        const int u = t - 8 * 32 * 16;
        const int e = u & 15, lane = (u >> 4) & 31, nt = (u >> 9) & 7, kt = u >> 12;
        const int k = kt * 32 + (lane >> 4) * 16 + e;
        const int c = nt * 16 + (lane & 15);
        wfrag[t] = (_Float16)W2[k * HDIM + c];
    }
}

// ---------------------------------------------------------------------------
// Kernel 1: per-supernode radius scan + K-nearest selection.
// One workgroup per supernode; pos (786 KB) stays resident in the 192 MB L2.
__global__ __launch_bounds__(256) void knn_kernel(const float* __restrict__ pos,
                                                  const int*   __restrict__ batch,
                                                  const int*   __restrict__ snidx,
                                                  const int*   __restrict__ snb,
                                                  int* __restrict__ nbr) {
    __shared__ float cd2[CAP];
    __shared__ int   cidx[CAP];
    __shared__ int   ccnt;

    const int m   = blockIdx.x;
    const int tid = threadIdx.x;
    if (tid == 0) ccnt = 0;
    if (tid < KNB) nbr[m * KNB + tid] = -1;

    const int   sn = snidx[m];
    const float sx = pos[sn * 3 + 0];
    const float sy = pos[sn * 3 + 1];
    const float sz = pos[sn * 3 + 2];
    const int   sb = snb[m];
    __syncthreads();

    for (int i = tid; i < N_PTS; i += 256) {
        if (i + 4096 < N_PTS) __builtin_prefetch(&pos[(i + 4096) * 3], 0, 0);
        const float dx = pos[i * 3 + 0] - sx;
        const float dy = pos[i * 3 + 1] - sy;
        const float dz = pos[i * 3 + 2] - sz;
        const float d2 = dx * dx + dy * dy + dz * dz;
        if (d2 <= R2C && batch[i] == sb) {
            const int slot = atomicAdd(&ccnt, 1);   // ds_add_rtn_u32
            if (slot < CAP) { cd2[slot] = d2; cidx[slot] = i; }
        }
    }
    __syncthreads();

    const int n = min(ccnt, CAP);
    if (tid < CAP) {
        const float myd = (tid < n) ? cd2[tid]  : 3.0e38f;
        const int   myi = (tid < n) ? cidx[tid] : -1;
        int rank = 0;
        for (int t = 0; t < n; ++t) {
            const float od = cd2[t];
            rank += (od < myd) || (od == myd && t < tid);
        }
        if (tid < n && rank < KNB) nbr[m * KNB + rank] = myi;
    }
}

// ---------------------------------------------------------------------------
// Kernel 2: WMMA message MLP + masked sum. One wave == one supernode.
__global__ __launch_bounds__(128) void mlp_kernel(const float* __restrict__ x,
                                                  const float* __restrict__ pos,
                                                  const _Float16* __restrict__ wfrag,
                                                  const float* __restrict__ b1,
                                                  const float* __restrict__ b2,
                                                  const int* __restrict__ snidx,
                                                  const int* __restrict__ nbr,
                                                  float* __restrict__ out) {
    // [0..255]   : W1 B-fragments (8 nt x 32 lanes)
    // [256..1279]: W2 B-fragments (4 kt x 8 nt x 32 lanes)
    __shared__ v16h              sWf[1280];          // 40 KB
    __shared__ v16h              sFeatV[WV][64];     // 8 KB  (32 rows x 32 halves per wave)
    __shared__ __align__(16) _Float16 sH[WV][32 * HDIM]; // 32 KB
    __shared__ float             sOut[WV][HDIM];     // 2 KB
    __shared__ int               sNbr[WV][KNB];      // 0.5 KB

    const int tid  = threadIdx.x;
    const int wave = tid >> 5;
    const int lane = tid & 31;
    const int half = lane >> 4;
    const int m    = blockIdx.x * WV + wave;

    // ---- stage all B-fragments (40 KB) into LDS via async copy ----
    {
        char*       dst = (char*)sWf;
        const char* src = (const char*)wfrag;
        for (int t = tid; t < 2560; t += 128)
            copy16_to_lds(dst + t * 16, src + t * 16);
        async_copy_fence();
    }

    // ---- per-wave neighbor list + validity ----
    const int idx = nbr[m * KNB + lane];
    sNbr[wave][lane] = idx;
    unsigned vmask;
#if defined(HAVE_BALLOT32)
    vmask = __builtin_amdgcn_ballot_w32(idx >= 0);
#else
    vmask = 0;
#pragma unroll
    for (int r = 0; r < KNB; ++r) vmask |= (sNbr[wave][r] >= 0) ? (1u << r) : 0u;
#endif
    const int nvalid = __popc(vmask);

    const int   sn = snidx[m];
    const float sx = pos[sn * 3 + 0];
    const float sy = pos[sn * 3 + 1];
    const float sz = pos[sn * 3 + 2];

    // ---- build feat row (lane r -> neighbor r): [x(16) | rel(3) | 0-pad] ----
    {
        v16h f0 = {}, f1 = {};
        if (idx >= 0) {
            const float4* xr = (const float4*)(x + (size_t)idx * DXF);
            const float4 a0 = xr[0], a1 = xr[1], a2 = xr[2], a3 = xr[3];
            f0[0]  = (_Float16)a0.x; f0[1]  = (_Float16)a0.y;
            f0[2]  = (_Float16)a0.z; f0[3]  = (_Float16)a0.w;
            f0[4]  = (_Float16)a1.x; f0[5]  = (_Float16)a1.y;
            f0[6]  = (_Float16)a1.z; f0[7]  = (_Float16)a1.w;
            f0[8]  = (_Float16)a2.x; f0[9]  = (_Float16)a2.y;
            f0[10] = (_Float16)a2.z; f0[11] = (_Float16)a2.w;
            f0[12] = (_Float16)a3.x; f0[13] = (_Float16)a3.y;
            f0[14] = (_Float16)a3.z; f0[15] = (_Float16)a3.w;
            f1[0]  = (_Float16)(pos[idx * 3 + 0] - sx);
            f1[1]  = (_Float16)(pos[idx * 3 + 1] - sy);
            f1[2]  = (_Float16)(pos[idx * 3 + 2] - sz);
        }
        sFeatV[wave][lane * 2 + 0] = f0;
        sFeatV[wave][lane * 2 + 1] = f1;
    }
    for (int c = lane; c < HDIM; c += 32) sOut[wave][c] = 0.0f;

    __syncthreads();   // sWf staged; feat/sH/sOut are wave-private & DS is in-order per wave

    // ---- GEMM1: h = gelu(feat[32x32] @ W1[32x128] + b1) -> sH (f16) ----
    for (int mt = 0; mt < 2; ++mt) {
        const v16h a =
            load_frag_a((const _Float16*)&sFeatV[wave][0] + mt * 16 * 32, 32, lane);
#pragma unroll
        for (int nt = 0; nt < 8; ++nt) {
            const v16h bf = sWf[nt * 32 + lane];       // single 32B LDS fragment load
            v8f c = {};
            c = __builtin_amdgcn_wmma_f32_16x16x32_f16(false, a, false, bf,
                                                       (short)0, c, false, false);
            const int   col  = nt * 16 + (lane & 15);
            const float bias = b1[col];
#pragma unroll
            for (int j = 0; j < 8; ++j) {
                const int row = mt * 16 + half * 8 + j;
                sH[wave][row * HDIM + col] = (_Float16)gelu_f(c[j] + bias);
            }
        }
    }

    // ---- GEMM2: y = h[32x128] @ W2[128x128]; masked row-sum -> sOut ----
    for (int mt = 0; mt < 2; ++mt) {
        const _Float16* hbase = &sH[wave][mt * 16 * HDIM];
        const v16h a0 = load_frag_a(hbase + 0,  HDIM, lane);
        const v16h a1 = load_frag_a(hbase + 32, HDIM, lane);
        const v16h a2 = load_frag_a(hbase + 64, HDIM, lane);
        const v16h a3 = load_frag_a(hbase + 96, HDIM, lane);
#pragma unroll
        for (int nt = 0; nt < 8; ++nt) {
            v8f c = {};
            c = __builtin_amdgcn_wmma_f32_16x16x32_f16(false, a0, false,
                    sWf[256 + (0 * 8 + nt) * 32 + lane], (short)0, c, false, false);
            c = __builtin_amdgcn_wmma_f32_16x16x32_f16(false, a1, false,
                    sWf[256 + (1 * 8 + nt) * 32 + lane], (short)0, c, false, false);
            c = __builtin_amdgcn_wmma_f32_16x16x32_f16(false, a2, false,
                    sWf[256 + (2 * 8 + nt) * 32 + lane], (short)0, c, false, false);
            c = __builtin_amdgcn_wmma_f32_16x16x32_f16(false, a3, false,
                    sWf[256 + (3 * 8 + nt) * 32 + lane], (short)0, c, false, false);
            const int col = nt * 16 + (lane & 15);
            float partial = 0.0f;
#pragma unroll
            for (int j = 0; j < 8; ++j) {
                const int row = mt * 16 + half * 8 + j;
                partial += ((vmask >> row) & 1u) ? c[j] : 0.0f;
            }
            atomicAdd(&sOut[wave][col], partial);   // ds_add_f32 (lane L vs L+16 share col)
        }
    }

    // out = masked sum + nvalid * b2
    for (int c = lane; c < HDIM; c += 32)
        out[(size_t)m * HDIM + c] = sOut[wave][c] + (float)nvalid * b2[c];
}

// ---------------------------------------------------------------------------
extern "C" void kernel_launch(void* const* d_in, const int* in_sizes, int n_in,
                              void* d_out, int out_size, void* d_ws, size_t ws_size,
                              hipStream_t stream) {
    (void)in_sizes; (void)n_in; (void)out_size; (void)ws_size;
    const float* x     = (const float*)d_in[0];
    const float* pos   = (const float*)d_in[1];
    const float* W1    = (const float*)d_in[2];
    const float* b1    = (const float*)d_in[3];
    const float* W2    = (const float*)d_in[4];
    const float* b2    = (const float*)d_in[5];
    const int*   batch = (const int*)d_in[6];
    const int*   snidx = (const int*)d_in[7];
    const int*   snb   = (const int*)d_in[8];
    float*       out   = (float*)d_out;

    char* ws = (char*)d_ws;
    int*      nbr   = (int*)ws;                                      // 256 KB
    _Float16* wfrag = (_Float16*)(ws + (size_t)M_SN * KNB * sizeof(int)); // 40 KB, 32B-aligned

    prep_weights<<<dim3(80),        dim3(256), 0, stream>>>(W1, W2, wfrag);
    knn_kernel  <<<dim3(M_SN),      dim3(256), 0, stream>>>(pos, batch, snidx, snb, nbr);
    mlp_kernel  <<<dim3(M_SN / WV), dim3(128), 0, stream>>>(x, pos, wfrag, b1, b2,
                                                            snidx, nbr, out);
}
